// NetworkAwareClassicalExpert_34136400069003
// MI455X (gfx1250) — matrix operand order repaired and checked by reading, not compiled, not run on hardware
//
#include <hip/hip_runtime.h>
#include <hip/hip_bf16.h>

// ---------------------------------------------------------------------------
// Types for CDNA5 WMMA (wave32): bf16 A/B fragments (16 halfs/lane = 8 VGPRs),
// f32 C/D fragment (8 VGPRs).
// ---------------------------------------------------------------------------
typedef __bf16 bf16;
typedef __attribute__((ext_vector_type(16))) __bf16 v16bf;
typedef __attribute__((ext_vector_type(8)))  __bf16 v8bf;
typedef __attribute__((ext_vector_type(8)))  float  v8f;

#define WB 256
#define WT 363
#define WC 180
#define WCD 96
#define WDM 128
#define WNH 4
#define WHD 32

static __device__ __forceinline__ float gelu_f(float x) {
    return 0.5f * x * (1.0f + erff(x * 0.70710678118654752f));
}
static __device__ __forceinline__ int imin_d(int a, int b) { return a < b ? a : b; }

// A-matrix (16x32 bf16): lane's 16 elements are the contiguous k range
// k = hi*16 + [0..15] (same order as B); m = lane&15.
static __device__ __forceinline__ v16bf load_a_frag(const bf16* __restrict__ row, int hi) {
    v8bf a0 = *(const v8bf*)(row + hi * 16);
    v8bf a1 = *(const v8bf*)(row + hi * 16 + 8);
    return __builtin_shufflevector(a0, a1, 0, 1, 2, 3, 4, 5, 6, 7,
                                   8, 9, 10, 11, 12, 13, 14, 15);
}
// B-matrix (32x16 bf16): lane&15 = N column, k = (lane>>4)*16 + e -> one
// aligned 32-byte contiguous region per lane.
// C/D (16x16 f32): n = lane&15, m = r + 8*(lane>>4).

// ---------------------------------------------------------------------------
// Generic batched WMMA GEMM:  C[m,n] = scale * sum_k A[m,k]*Bt[n,k] + bias[n]
// A: M x K bf16 row-major (lda), M must be a multiple of 16 (pad with zeros).
// Bt: N x K bf16 row-major (ldb), N multiple of 32, K multiple of 32.
// Each wave computes a 16x32 tile (2 WMMAs/K-step sharing the A fragment).
// Epilogue act: 0=none, 1=gelu, 2=relu. Optional f32 and/or bf16 outputs.
// ---------------------------------------------------------------------------
__global__ __launch_bounds__(128) void wmma_gemm_kernel(
    const bf16* __restrict__ A, int lda, long long bsA,
    const bf16* __restrict__ Bt, int ldb, long long bsB,
    const float* __restrict__ bias, float scale,
    float* __restrict__ Cf, bf16* __restrict__ Cb, int ldc, long long bsC,
    int M, int N, int K, int act)
{
    const int Nt2 = N >> 5;               // 32-wide N tiles
    const int Mt  = M >> 4;
    const int tile = blockIdx.x * 4 + (threadIdx.x >> 5);
    if (tile >= Mt * Nt2) return;
    const int mt = tile / Nt2, nt = tile % Nt2;
    const int lane = threadIdx.x & 31;
    const int mloc = lane & 15;
    const int hi   = lane >> 4;
    const int z = blockIdx.z;

    const bf16* Ab  = A  + (size_t)z * (size_t)bsA;
    const bf16* Btb = Bt + (size_t)z * (size_t)bsB;

    const bf16* arow  = Ab  + (size_t)(mt * 16 + mloc) * (size_t)lda;
    const bf16* bcol0 = Btb + (size_t)(nt * 32 + mloc) * (size_t)ldb + hi * 16;
    const bf16* bcol1 = bcol0 + (size_t)16 * (size_t)ldb;

    v8f acc0, acc1;
    #pragma unroll
    for (int r = 0; r < 8; ++r) { acc0[r] = 0.0f; acc1[r] = 0.0f; }

    for (int k0 = 0; k0 < K; k0 += 32) {
        v16bf af = load_a_frag(arow + k0, hi);
        v16bf b0 = *(const v16bf*)(bcol0 + k0);
        v16bf b1 = *(const v16bf*)(bcol1 + k0);
        acc0 = __builtin_amdgcn_wmma_f32_16x16x32_bf16(false, af, false, b0,
                                                       (short)0, acc0, false, false);
        acc1 = __builtin_amdgcn_wmma_f32_16x16x32_bf16(false, af, false, b1,
                                                       (short)0, acc1, false, false);
    }

    #pragma unroll
    for (int r = 0; r < 8; ++r) {
        const int mm = mt * 16 + r + hi * 8;
        const int n0 = nt * 32 + mloc;
        float v0 = acc0[r] * scale + (bias ? bias[n0] : 0.0f);
        float v1 = acc1[r] * scale + (bias ? bias[n0 + 16] : 0.0f);
        if (act == 1)      { v0 = gelu_f(v0);     v1 = gelu_f(v1); }
        else if (act == 2) { v0 = fmaxf(v0, 0.f); v1 = fmaxf(v1, 0.f); }
        const size_t idx = (size_t)z * (size_t)bsC + (size_t)mm * (size_t)ldc + n0;
        if (Cf) { Cf[idx] = v0; Cf[idx + 16] = v1; }
        if (Cb) { Cb[idx] = (bf16)v0; Cb[idx + 16] = (bf16)v1; }
    }
}

// ---------------------------------------------------------------------------
// Stage 1: per-net projection + gelu.  h[bt, n*8+o] over 96 channels.
// ---------------------------------------------------------------------------
__global__ void proj_kernel(const float* __restrict__ x,
                            const float* __restrict__ w_proj,
                            const float* __restrict__ b_proj,
                            bf16* __restrict__ hbf, int total)
{
    int idx = blockIdx.x * blockDim.x + threadIdx.x;
    if (idx >= total) return;
    int c  = idx % 96;
    size_t bt = (size_t)(idx / 96);
    int n = c >> 3, o = c & 7;
    const float* xr = x + bt * 180 + n * 15;
    float acc = b_proj[n * 8 + o];
    #pragma unroll
    for (int i = 0; i < 15; ++i)
        acc += xr[i] * w_proj[(n * 15 + i) * 8 + o];
    hbf[idx] = (bf16)gelu_f(acc);
}

// ---------------------------------------------------------------------------
// Dilated depthwise conv (7 taps) along T, per branch k, dilation d.
// ---------------------------------------------------------------------------
__global__ void dwconv_kernel(const bf16* __restrict__ hbf,
                              const float* __restrict__ dw_w,
                              const float* __restrict__ dw_b,
                              bf16* __restrict__ ybf, int k, int d, int total)
{
    int idx = blockIdx.x * blockDim.x + threadIdx.x;
    if (idx >= total) return;
    int c = idx % 96;
    int bt = idx / 96;
    int b = bt / WT, t = bt % WT;
    float acc = dw_b[k * 96 + c];
    const float* wr = dw_w + (size_t)(k * 96 + c) * 7;
    #pragma unroll
    for (int j = 0; j < 7; ++j) {
        int tt = t + (j - 3) * d;
        if (tt >= 0 && tt < WT)
            acc += wr[j] * (float)hbf[((size_t)b * WT + tt) * 96 + c];
    }
    ybf[idx] = (bf16)acc;
}

// ---------------------------------------------------------------------------
// GroupNorm stats per (b, group): mean/rstd over 12 channels x T.
// ---------------------------------------------------------------------------
__global__ __launch_bounds__(256) void gn_stats_kernel(const float* __restrict__ zf,
                                                       float* __restrict__ stats)
{
    __shared__ float rs[256], rs2[256];
    int bg = blockIdx.x;
    int b = bg >> 3, g = bg & 7;
    float s = 0.0f, s2 = 0.0f;
    for (int i = threadIdx.x; i < 12 * WT; i += 256) {
        int ci = i % 12, t = i / 12;
        float v = zf[((size_t)b * WT + t) * 96 + g * 12 + ci];
        s += v; s2 += v * v;
    }
    rs[threadIdx.x] = s; rs2[threadIdx.x] = s2;
    __syncthreads();
    for (int off = 128; off > 0; off >>= 1) {
        if ((int)threadIdx.x < off) { rs[threadIdx.x] += rs[threadIdx.x + off];
                                      rs2[threadIdx.x] += rs2[threadIdx.x + off]; }
        __syncthreads();
    }
    if (threadIdx.x == 0) {
        float inv = 1.0f / (12.0f * WT);
        float mean = rs[0] * inv;
        float var = rs2[0] * inv - mean * mean;
        stats[bg * 2 + 0] = mean;
        stats[bg * 2 + 1] = rsqrtf(var + 1e-5f);
    }
}

__global__ void gn_norm_kernel(const float* __restrict__ zf,
                               const float* __restrict__ stats,
                               const float* __restrict__ gn_g,
                               const float* __restrict__ gn_b,
                               bf16* __restrict__ catbf, int k, int total)
{
    int idx = blockIdx.x * blockDim.x + threadIdx.x;
    if (idx >= total) return;
    int c = idx % 96;
    int bt = idx / 96;
    int b = bt / WT;
    int g = c / 12;
    float mean = stats[(b * 8 + g) * 2 + 0];
    float rstd = stats[(b * 8 + g) * 2 + 1];
    float v = (zf[idx] - mean) * rstd * gn_g[k * 96 + c] + gn_b[k * 96 + c];
    catbf[(size_t)bt * 288 + k * 96 + c] = (bf16)gelu_f(v);
}

// ---------------------------------------------------------------------------
// Row LayerNorm over 128 cols (+optional residual, +optional gelu).
// One wave (32 lanes) per row, 4 lanes-strided cols each; wave32 shuffles.
// ---------------------------------------------------------------------------
__global__ __launch_bounds__(128) void ln_kernel(const float* __restrict__ X,
                                                 const float* __restrict__ R,
                                                 const float* __restrict__ g,
                                                 const float* __restrict__ bta,
                                                 float* __restrict__ outF,
                                                 bf16* __restrict__ outB,
                                                 int rows, int act)
{
    int row = blockIdx.x * 4 + (threadIdx.x >> 5);
    if (row >= rows) return;
    int lane = threadIdx.x & 31;
    float v[4];
    #pragma unroll
    for (int j = 0; j < 4; ++j) {
        int c = lane + j * 32;
        v[j] = X[(size_t)row * 128 + c] + (R ? R[(size_t)row * 128 + c] : 0.0f);
    }
    float s = v[0] + v[1] + v[2] + v[3];
    #pragma unroll
    for (int off = 16; off > 0; off >>= 1) s += __shfl_xor(s, off, 32);
    float mean = s * (1.0f / 128.0f);
    float q = 0.0f;
    #pragma unroll
    for (int j = 0; j < 4; ++j) { float d = v[j] - mean; q += d * d; }
    #pragma unroll
    for (int off = 16; off > 0; off >>= 1) q += __shfl_xor(q, off, 32);
    float rstd = rsqrtf(q * (1.0f / 128.0f) + 1e-5f);
    #pragma unroll
    for (int j = 0; j < 4; ++j) {
        int c = lane + j * 32;
        float o = (v[j] - mean) * rstd * g[c] + bta[c];
        if (act == 1) o = gelu_f(o);
        if (outF) outF[(size_t)row * 128 + c] = o;
        if (outB) outB[(size_t)row * 128 + c] = (bf16)o;
    }
}

// ---------------------------------------------------------------------------
// Flash attention, one (b,h) per 128-thread block (4 waves), wave owns 16
// query rows, streams 32 keys per step.  qkv: [B*T, 384] bf16 (q|k|v).
// Scores via v_wmma_f32_16x16x32_bf16 (HD=32 == one K step); online softmax
// in per-wave LDS; P*V via two more WMMAs.
// ---------------------------------------------------------------------------
__global__ __launch_bounds__(128) void attn_kernel(const bf16* __restrict__ qkv,
                                                   bf16* __restrict__ outp)
{
    __shared__ float sc_all[4][16 * 32];
    __shared__ float al_all[4][16];
    __shared__ float li_all[4][16];
    const int bh = blockIdx.x;
    const int b = bh / WNH, h = bh % WNH;
    const int wave = threadIdx.x >> 5;
    const int lane = threadIdx.x & 31;
    const int mloc = lane & 15, hi = lane >> 4;
    float* sc = sc_all[wave];
    float* al = al_all[wave];
    float* li = li_all[wave];
    const float scale = 0.17677669529663687f; // 1/sqrt(32)
    const size_t rowstride = 384;

    for (int qb = wave; qb * 16 < WT; qb += 4) {
        // load Q fragment (rows qb*16..+15, 32 dims) - two aligned b128 loads
        int tq = qb * 16 + mloc;
        int tqc = imin_d(tq, WT - 1);
        const bf16* qrow = qkv + ((size_t)b * WT + tqc) * rowstride + h * 32;
        v16bf aq = load_a_frag(qrow, hi);

        v8f o0, o1;
        #pragma unroll
        for (int r = 0; r < 8; ++r) { o0[r] = 0.0f; o1[r] = 0.0f; }
        float m_i = -1e30f, l_i = 0.0f;   // owned by lanes 0..15 (row = lane)

        for (int kb = 0; kb * 32 < WT; ++kb) {
            const int s0 = kb * 32;
            // K^T fragments: B[k][n] = K[s0(+16)+n][k]; 32B contiguous per lane
            const int sA = imin_d(s0 + mloc, WT - 1);
            const int sB = imin_d(s0 + 16 + mloc, WT - 1);
            v16bf bk0 = *(const v16bf*)(qkv + ((size_t)b * WT + sA) * rowstride
                                        + 128 + h * 32 + hi * 16);
            v16bf bk1 = *(const v16bf*)(qkv + ((size_t)b * WT + sB) * rowstride
                                        + 128 + h * 32 + hi * 16);
            v8f z8;
            #pragma unroll
            for (int r = 0; r < 8; ++r) z8[r] = 0.0f;
            v8f s0f = __builtin_amdgcn_wmma_f32_16x16x32_bf16(false, aq, false, bk0,
                                                              (short)0, z8, false, false);
            v8f s1f = __builtin_amdgcn_wmma_f32_16x16x32_bf16(false, aq, false, bk1,
                                                              (short)0, z8, false, false);
            // stage scaled+masked scores into per-wave LDS
            #pragma unroll
            for (int r = 0; r < 8; ++r) {
                int mm = r + hi * 8;
                int key0 = s0 + mloc, key1 = s0 + 16 + mloc;
                sc[mm * 32 + mloc]      = (key0 < WT) ? s0f[r] * scale : -1e30f;
                sc[mm * 32 + 16 + mloc] = (key1 < WT) ? s1f[r] * scale : -1e30f;
            }
            __builtin_amdgcn_wave_barrier();
            // online softmax, one row per lane (lanes 0..15)
            if (lane < 16) {
                float rmax = -1e30f;
                for (int j = 0; j < 32; ++j) rmax = fmaxf(rmax, sc[lane * 32 + j]);
                float mnew = fmaxf(m_i, rmax);
                float alpha = __expf(m_i - mnew);
                float rsum = 0.0f;
                for (int j = 0; j < 32; ++j) {
                    float p = __expf(sc[lane * 32 + j] - mnew);
                    sc[lane * 32 + j] = p;
                    rsum += p;
                }
                l_i = l_i * alpha + rsum;
                m_i = mnew;
                al[lane] = alpha;
            }
            __builtin_amdgcn_wave_barrier();
            // rescale running O by alpha
            #pragma unroll
            for (int r = 0; r < 8; ++r) {
                float a_ = al[r + hi * 8];
                o0[r] *= a_; o1[r] *= a_;
            }
            // P fragment from LDS (two contiguous 8-float chunks per lane)
            v16bf ap;
            #pragma unroll
            for (int e = 0; e < 8; ++e) {
                ap[e]     = (bf16)sc[mloc * 32 + hi * 16 + e];
                ap[8 + e] = (bf16)sc[mloc * 32 + hi * 16 + 8 + e];
            }
            // V fragments (transposed access, per-element)
            const bf16* vbase = qkv + (size_t)b * WT * rowstride + 256 + h * 32;
            v16bf bv0, bv1;
            #pragma unroll
            for (int e = 0; e < 16; ++e) {
                int s = imin_d(s0 + hi * 16 + e, WT - 1);
                bv0[e] = vbase[(size_t)s * rowstride + mloc];
                bv1[e] = vbase[(size_t)s * rowstride + 16 + mloc];
            }
            o0 = __builtin_amdgcn_wmma_f32_16x16x32_bf16(false, ap, false, bv0,
                                                         (short)0, o0, false, false);
            o1 = __builtin_amdgcn_wmma_f32_16x16x32_bf16(false, ap, false, bv1,
                                                         (short)0, o1, false, false);
            __builtin_amdgcn_wave_barrier();
        }
        if (lane < 16) li[lane] = 1.0f / l_i;
        __builtin_amdgcn_wave_barrier();
        #pragma unroll
        for (int r = 0; r < 8; ++r) {
            int mm = r + hi * 8;
            int t = qb * 16 + mm;
            if (t < WT) {
                float inv = li[mm];
                size_t base = ((size_t)b * WT + t) * 128 + h * 32;
                outp[base + mloc]      = (bf16)(o0[r] * inv);
                outp[base + 16 + mloc] = (bf16)(o1[r] * inv);
            }
        }
        __builtin_amdgcn_wave_barrier();
    }
}

// ---------------------------------------------------------------------------
// Attention pooling: w = softmax(h@pool_w + b) over T, temporal = sum w*h.
// ---------------------------------------------------------------------------
__global__ __launch_bounds__(256) void pool_kernel(const float* __restrict__ h4,
                                                   const float* __restrict__ pool_w,
                                                   const float* __restrict__ pool_b,
                                                   float* __restrict__ temporal)
{
    __shared__ float lw[WT];
    __shared__ float red[256];
    const int b = blockIdx.x;
    const int tid = threadIdx.x;
    float lmax = -1e30f;
    for (int t = tid; t < WT; t += 256) {
        const float* hr = h4 + ((size_t)b * WT + t) * 128;
        float acc = pool_b[0];
        for (int c = 0; c < 128; ++c) acc += hr[c] * pool_w[c];
        lw[t] = acc;
        lmax = fmaxf(lmax, acc);
    }
    red[tid] = lmax; __syncthreads();
    for (int off = 128; off > 0; off >>= 1) {
        if (tid < off) red[tid] = fmaxf(red[tid], red[tid + off]);
        __syncthreads();
    }
    const float gmax = red[0];
    __syncthreads();
    float lsum = 0.0f;
    for (int t = tid; t < WT; t += 256) {
        float e = __expf(lw[t] - gmax);
        lw[t] = e;
        lsum += e;
    }
    red[tid] = lsum; __syncthreads();
    for (int off = 128; off > 0; off >>= 1) {
        if (tid < off) red[tid] += red[tid + off];
        __syncthreads();
    }
    const float inv = 1.0f / red[0];
    __syncthreads();
    for (int c = tid; c < 128; c += 256) {
        float acc = 0.0f;
        for (int t = 0; t < WT; ++t)
            acc += lw[t] * h4[((size_t)b * WT + t) * 128 + c];
        temporal[(size_t)b * 128 + c] = acc * inv;
    }
}

// ---------------------------------------------------------------------------
// Correlation normalize: xnT[b][c][t] (padded to 192 rows x 384 cols, bf16).
// ---------------------------------------------------------------------------
__global__ __launch_bounds__(192) void xnorm_kernel(const float* __restrict__ x,
                                                    bf16* __restrict__ xnT)
{
    const int b = blockIdx.x;
    const int c = threadIdx.x;
    bf16* dst = xnT + ((size_t)b * 192 + c) * 384;
    if (c >= WC) {
        for (int t = 0; t < 384; ++t) dst[t] = (bf16)0.0f;
        return;
    }
    const float* xc = x + (size_t)b * WT * WC + c;
    float s = 0.0f;
    for (int t = 0; t < WT; ++t) s += xc[(size_t)t * WC];
    float mean = s / (float)WT;
    float q = 0.0f;
    for (int t = 0; t < WT; ++t) { float d = xc[(size_t)t * WC] - mean; q += d * d; }
    float stdv = sqrtf(q / (float)(WT - 1));
    float r = 1.0f / fmaxf(stdv, 1e-8f);
    for (int t = 0; t < WT; ++t) dst[t] = (bf16)((xc[(size_t)t * WC] - mean) * r);
    for (int t = WT; t < 384; ++t) dst[t] = (bf16)0.0f;
}

// ---------------------------------------------------------------------------
// Block-mean of corr (192-padded) over 15x15, triu(12) gather, fc 78->64+gelu.
// ---------------------------------------------------------------------------
__global__ __launch_bounds__(128) void bmfc_kernel(const float* __restrict__ corr,
                                                   const float* __restrict__ fc_w,
                                                   const float* __restrict__ fc_b,
                                                   float* __restrict__ fcbuf)
{
    __shared__ float fcv[78];
    const int b = blockIdx.x;
    const int tid = threadIdx.x;
    if (tid < 78) {
        int i = 0, rem = tid;
        while (rem >= 12 - i) { rem -= 12 - i; ++i; }
        int j = i + rem;
        const float* cb = corr + (size_t)b * 192 * 192;
        float s = 0.0f;
        for (int a = 0; a < 15; ++a)
            for (int d = 0; d < 15; ++d)
                s += cb[(size_t)(i * 15 + a) * 192 + (j * 15 + d)];
        fcv[tid] = s * (1.0f / 225.0f);
    }
    __syncthreads();
    if (tid < 64) {
        float acc = fc_b[tid];
        for (int i = 0; i < 78; ++i) acc += fcv[i] * fc_w[i * 64 + tid];
        fcbuf[(size_t)b * 64 + tid] = gelu_f(acc);
    }
}

// ---------------------------------------------------------------------------
// Fusion head: concat(temporal, fc) -> fus1 -> LN -> gelu -> fus2 -> out.
// ---------------------------------------------------------------------------
__global__ __launch_bounds__(128) void fuse_kernel(const float* __restrict__ temporal,
                                                   const float* __restrict__ fcbuf,
                                                   const float* __restrict__ fus1_w,
                                                   const float* __restrict__ fus1_b,
                                                   const float* __restrict__ ln_g,
                                                   const float* __restrict__ ln_b,
                                                   const float* __restrict__ fus2_w,
                                                   const float* __restrict__ fus2_b,
                                                   float* __restrict__ out)
{
    __shared__ float zs[128];
    __shared__ float red[128];
    const int b = blockIdx.x;
    const int j = threadIdx.x;
    float acc = fus1_b[j];
    for (int i = 0; i < 128; ++i) acc += temporal[(size_t)b * 128 + i] * fus1_w[i * 128 + j];
    for (int i = 0; i < 64; ++i)  acc += fcbuf[(size_t)b * 64 + i] * fus1_w[(128 + i) * 128 + j];
    red[j] = acc; __syncthreads();
    for (int off = 64; off > 0; off >>= 1) {
        if (j < off) red[j] += red[j + off];
        __syncthreads();
    }
    const float mean = red[0] * (1.0f / 128.0f);
    __syncthreads();
    float d = acc - mean;
    red[j] = d * d; __syncthreads();
    for (int off = 64; off > 0; off >>= 1) {
        if (j < off) red[j] += red[j + off];
        __syncthreads();
    }
    const float rstd = rsqrtf(red[0] * (1.0f / 128.0f) + 1e-5f);
    __syncthreads();
    zs[j] = gelu_f((acc - mean) * rstd * ln_g[j] + ln_b[j]);
    __syncthreads();
    if (j < 64) {
        float o = fus2_b[j];
        for (int q = 0; q < 128; ++q) o += zs[q] * fus2_w[q * 64 + j];
        out[(size_t)b * 64 + j] = o;
    }
}

// ---------------------------------------------------------------------------
// Weight conversion helpers (f32 -> bf16, optionally transposed).
// ---------------------------------------------------------------------------
__global__ void cvt_bf16_kernel(const float* __restrict__ s, bf16* __restrict__ d, int n)
{
    int i = blockIdx.x * blockDim.x + threadIdx.x;
    if (i < n) d[i] = (bf16)s[i];
}
__global__ void transpose_bf16_kernel(const float* __restrict__ s, bf16* __restrict__ d,
                                      int R, int C)
{
    int i = blockIdx.x * blockDim.x + threadIdx.x;
    if (i < R * C) {
        int r = i / C, c = i % C;
        d[(size_t)c * R + r] = (bf16)s[i];
    }
}

// ===========================================================================
// Host orchestration
// ===========================================================================
static inline unsigned ceil_div_h(size_t a, size_t b) { return (unsigned)((a + b - 1) / b); }

extern "C" void kernel_launch(void* const* d_in, const int* in_sizes, int n_in,
                              void* d_out, int out_size, void* d_ws, size_t ws_size,
                              hipStream_t stream) {
    (void)in_sizes; (void)n_in; (void)out_size; (void)ws_size;
    const float* x          = (const float*)d_in[0];
    const float* w_proj     = (const float*)d_in[1];
    const float* b_proj     = (const float*)d_in[2];
    const float* dw_w       = (const float*)d_in[3];
    const float* dw_b       = (const float*)d_in[4];
    const float* pw_w       = (const float*)d_in[5];
    const float* pw_b       = (const float*)d_in[6];
    const float* gn_g       = (const float*)d_in[7];
    const float* gn_b       = (const float*)d_in[8];
    const float* merge_w    = (const float*)d_in[9];
    const float* merge_b    = (const float*)d_in[10];
    const float* merge_ln_g = (const float*)d_in[11];
    const float* merge_ln_b = (const float*)d_in[12];
    const float* qkv_w      = (const float*)d_in[13];
    const float* qkv_b      = (const float*)d_in[14];
    const float* out_w      = (const float*)d_in[15];
    const float* out_b      = (const float*)d_in[16];
    const float* ln1_g      = (const float*)d_in[17];
    const float* ln1_b      = (const float*)d_in[18];
    const float* ff1_w      = (const float*)d_in[19];
    const float* ff1_b      = (const float*)d_in[20];
    const float* ff2_w      = (const float*)d_in[21];
    const float* ff2_b      = (const float*)d_in[22];
    const float* ln2_g      = (const float*)d_in[23];
    const float* ln2_b      = (const float*)d_in[24];
    const float* pool_w     = (const float*)d_in[25];
    const float* pool_b     = (const float*)d_in[26];
    const float* fc_w       = (const float*)d_in[27];
    const float* fc_b       = (const float*)d_in[28];
    const float* fus1_w     = (const float*)d_in[29];
    const float* fus1_b     = (const float*)d_in[30];
    const float* fus_ln_g   = (const float*)d_in[31];
    const float* fus_ln_b   = (const float*)d_in[32];
    const float* fus2_w     = (const float*)d_in[33];
    const float* fus2_b     = (const float*)d_in[34];
    float* outp = (float*)d_out;

    const size_t NT = (size_t)WB * WT;            // 92928 (divisible by 16)
    char* w = (char*)d_ws;

    // ---- workspace slabs (reused across phases) ----
    const size_t o_R0 = 0;                        // NT*768 B
    const size_t o_R1 = o_R0 + NT * 768;          // NT*576 B
    const size_t o_R2 = o_R1 + NT * 576;          // NT*512 B
    const size_t o_R3 = o_R2 + NT * 512;          // NT*512 B
    const size_t o_R4 = o_R3 + NT * 512;          // NT*256 B
    const size_t o_R5 = o_R4 + NT * 256;          // NT*512 B
    size_t o_sm = o_R5 + NT * 512;

    float* gstats   = (float*)(w + o_sm); o_sm += (size_t)WB * 8 * 2 * 4;
    float* temporal = (float*)(w + o_sm); o_sm += (size_t)WB * 128 * 4;
    float* fcbuf    = (float*)(w + o_sm); o_sm += (size_t)WB * 64 * 4;
    bf16* mergeT = (bf16*)(w + o_sm); o_sm += (size_t)128 * 288 * 2;
    bf16* qkvW   = (bf16*)(w + o_sm); o_sm += (size_t)384 * 128 * 2;
    bf16* outW   = (bf16*)(w + o_sm); o_sm += (size_t)128 * 128 * 2;
    bf16* ff1W   = (bf16*)(w + o_sm); o_sm += (size_t)256 * 128 * 2;
    bf16* ff2W   = (bf16*)(w + o_sm); o_sm += (size_t)128 * 256 * 2;
    bf16* pwW    = (bf16*)(w + o_sm); o_sm += (size_t)3 * 96 * 96 * 2;

    // phase aliases
    bf16*  hbf     = (bf16*)(w + o_R0);
    bf16*  ybf     = (bf16*)(w + o_R0 + NT * 192);
    float* zf      = (float*)(w + o_R0 + NT * 384);
    bf16*  qkvbf   = (bf16*)(w + o_R0);
    bf16*  ffmid   = (bf16*)(w + o_R0);
    bf16*  xnT     = (bf16*)(w + o_R0);
    bf16*  catbf   = (bf16*)(w + o_R1);
    bf16*  oattnbf = (bf16*)(w + o_R1);
    float* corr    = (float*)(w + o_R1);
    float* h2raw   = (float*)(w + o_R2);
    float* oproj   = (float*)(w + o_R2);
    float* ff2out  = (float*)(w + o_R2);
    float* h2f     = (float*)(w + o_R3);
    float* h4f     = (float*)(w + o_R3);
    bf16*  h2bf    = (bf16*)(w + o_R4);
    bf16*  h3bf    = (bf16*)(w + o_R4);
    float* h3f     = (float*)(w + o_R5);

    // ---- 0) convert weights to bf16 (weight-major N x K layout) ----
    transpose_bf16_kernel<<<ceil_div_h(288 * 128, 256), 256, 0, stream>>>(merge_w, mergeT, 288, 128);
    cvt_bf16_kernel<<<ceil_div_h(384 * 128, 256), 256, 0, stream>>>(qkv_w, qkvW, 384 * 128);
    cvt_bf16_kernel<<<ceil_div_h(128 * 128, 256), 256, 0, stream>>>(out_w, outW, 128 * 128);
    cvt_bf16_kernel<<<ceil_div_h(256 * 128, 256), 256, 0, stream>>>(ff1_w, ff1W, 256 * 128);
    cvt_bf16_kernel<<<ceil_div_h(128 * 256, 256), 256, 0, stream>>>(ff2_w, ff2W, 128 * 256);
    cvt_bf16_kernel<<<ceil_div_h(3 * 96 * 96, 256), 256, 0, stream>>>(pw_w, pwW, 3 * 96 * 96);

    // ---- 1) projection + gelu -> hbf [NT,96] ----
    proj_kernel<<<ceil_div_h(NT * 96, 256), 256, 0, stream>>>(x, w_proj, b_proj, hbf, (int)(NT * 96));

    // ---- 2) three dilated conv branches ----
    const int dils[3] = {1, 4, 16};
    for (int k = 0; k < 3; ++k) {
        dwconv_kernel<<<ceil_div_h(NT * 96, 256), 256, 0, stream>>>(
            hbf, dw_w, dw_b, ybf, k, dils[k], (int)(NT * 96));
        {   // pointwise 96x96 GEMM (WMMA), f32 out
            int M = (int)NT, N = 96, K = 96;
            int tiles = (M / 16) * (N / 32);
            dim3 grid(ceil_div_h(tiles, 4), 1, 1);
            wmma_gemm_kernel<<<grid, 128, 0, stream>>>(
                ybf, 96, 0, pwW + (size_t)k * 96 * 96, 96, 0,
                pw_b + k * 96, 1.0f, zf, (bf16*)nullptr, 96, 0, M, N, K, 0);
        }
        gn_stats_kernel<<<WB * 8, 256, 0, stream>>>(zf, gstats);
        gn_norm_kernel<<<ceil_div_h(NT * 96, 256), 256, 0, stream>>>(
            zf, gstats, gn_g, gn_b, catbf, k, (int)(NT * 96));
    }

    // ---- 3) merge 288->128 (WMMA) + LN + gelu ----
    {
        int M = (int)NT, N = 128, K = 288;
        dim3 grid(ceil_div_h((M / 16) * (N / 32), 4), 1, 1);
        wmma_gemm_kernel<<<grid, 128, 0, stream>>>(
            catbf, 288, 0, mergeT, 288, 0, merge_b, 1.0f,
            h2raw, (bf16*)nullptr, 128, 0, M, N, K, 0);
    }
    ln_kernel<<<ceil_div_h(NT, 4), 128, 0, stream>>>(
        h2raw, (const float*)nullptr, merge_ln_g, merge_ln_b, h2f, h2bf, (int)NT, 1);

    // ---- 4) qkv 128->384 (WMMA, bf16 out) ----
    {
        int M = (int)NT, N = 384, K = 128;
        dim3 grid(ceil_div_h((M / 16) * (N / 32), 4), 1, 1);
        wmma_gemm_kernel<<<grid, 128, 0, stream>>>(
            h2bf, 128, 0, qkvW, 128, 0, qkv_b, 1.0f,
            (float*)nullptr, qkvbf, 384, 0, M, N, K, 0);
    }

    // ---- 5) flash attention (WMMA) -> oattnbf [NT,128] bf16 ----
    attn_kernel<<<WB * WNH, 128, 0, stream>>>(qkvbf, oattnbf);

    // ---- 6) output projection + residual LN ----
    {
        int M = (int)NT, N = 128, K = 128;
        dim3 grid(ceil_div_h((M / 16) * (N / 32), 4), 1, 1);
        wmma_gemm_kernel<<<grid, 128, 0, stream>>>(
            oattnbf, 128, 0, outW, 128, 0, out_b, 1.0f,
            oproj, (bf16*)nullptr, 128, 0, M, N, K, 0);
    }
    ln_kernel<<<ceil_div_h(NT, 4), 128, 0, stream>>>(
        oproj, h2f, ln1_g, ln1_b, h3f, h3bf, (int)NT, 0);

    // ---- 7) FFN: 128->256 relu (bf16 out), 256->128, residual LN ----
    {
        int M = (int)NT, N = 256, K = 128;
        dim3 grid(ceil_div_h((M / 16) * (N / 32), 4), 1, 1);
        wmma_gemm_kernel<<<grid, 128, 0, stream>>>(
            h3bf, 128, 0, ff1W, 128, 0, ff1_b, 1.0f,
            (float*)nullptr, ffmid, 256, 0, M, N, K, 2);
    }
    {
        int M = (int)NT, N = 128, K = 256;
        dim3 grid(ceil_div_h((M / 16) * (N / 32), 4), 1, 1);
        wmma_gemm_kernel<<<grid, 128, 0, stream>>>(
            ffmid, 256, 0, ff2W, 256, 0, ff2_b, 1.0f,
            ff2out, (bf16*)nullptr, 128, 0, M, N, K, 0);
    }
    ln_kernel<<<ceil_div_h(NT, 4), 128, 0, stream>>>(
        ff2out, h3f, ln2_g, ln2_b, h4f, (bf16*)nullptr, (int)NT, 0);

    // ---- 8) attention pooling -> temporal [B,128] ----
    pool_kernel<<<WB, 256, 0, stream>>>(h4f, pool_w, pool_b, temporal);

    // ---- 9) correlation branch: normalize, batched WMMA GEMM, bm+fc ----
    xnorm_kernel<<<WB, 192, 0, stream>>>(x, xnT);
    {
        int M = 192, N = 192, K = 384;
        long long bs = (long long)192 * 384;
        long long bc = (long long)192 * 192;
        dim3 grid(ceil_div_h((M / 16) * (N / 32), 4), 1, WB);
        wmma_gemm_kernel<<<grid, 128, 0, stream>>>(
            xnT, 384, bs, xnT, 384, bs, (const float*)nullptr,
            1.0f / (float)(WT - 1), corr, (bf16*)nullptr, 192, bc, M, N, K, 0);
    }
    bmfc_kernel<<<WB, 128, 0, stream>>>(corr, fc_w, fc_b, fcbuf);

    // ---- 10) fusion head -> d_out [B,64] ----
    fuse_kernel<<<WB, 128, 0, stream>>>(temporal, fcbuf, fus1_w, fus1_b,
                                        fus_ln_g, fus_ln_b, fus2_w, fus2_b, outp);
}
